// StyleGAN2Generator_31662498906236
// MI455X (gfx1250) — matrix operand order repaired and checked by reading, not compiled, not run on hardware
//
#include <hip/hip_runtime.h>
#include <hip/hip_bf16.h>

#define B_    4
#define CIN_  256
#define COUT_ 128
#define H_    64
#define W_    64
#define SDIM_ 512
#define OH_   128
#define OW_   128

#define PROWS 66                  // padded NHWC rows (H + 2)
#define PCOLS 66                  // padded NHWC cols (W + 2)

#define XROWS   3
#define XCOLS   66
#define NROWS   (XROWS * XCOLS)   // 198 staged pixels per chunk
#define CICHUNK 32
#define NCHUNK  (CIN_ / CICHUNK)  // 8
#define XSBUF   (NROWS * CICHUNK) // elements per LDS buffer

typedef __attribute__((ext_vector_type(16))) __bf16       v16bf;
typedef __attribute__((ext_vector_type(8)))  float        v8f;
typedef __attribute__((ext_vector_type(4)))  unsigned int u32x4;
typedef __attribute__((ext_vector_type(8)))  unsigned int u32x8;
typedef __attribute__((ext_vector_type(8)))  int          i32x8;
typedef __attribute__((ext_vector_type(4)))  int          i32x4;

struct U4x2 { u32x4 lo; u32x4 hi; };
union AFrag { v16bf bf; U4x2 q; u32x8 u; };

#if defined(__gfx1250__) && __has_builtin(__builtin_amdgcn_tensor_load_to_lds) && \
    __has_builtin(__builtin_amdgcn_s_wait_tensorcnt)
#define HAVE_TDM 1
#endif

__device__ __forceinline__ unsigned short f2bf(float f) {
  union { float f; unsigned int u; } v; v.f = f;
  unsigned int r = v.u + 0x7FFFu + ((v.u >> 16) & 1u);
  return (unsigned short)(r >> 16);
}

// ---------------- prep 1: s[b][ci] = style @ mod_w^T / sqrt(512) + mod_b
__global__ void k_style(const float* __restrict__ style, const float* __restrict__ mod_w,
                        const float* __restrict__ mod_b, float* __restrict__ s) {
  int t = blockIdx.x * blockDim.x + threadIdx.x;
  if (t >= B_ * CIN_) return;
  int b = t / CIN_, ci = t % CIN_;
  const float smul = 0.044194173824159216f; // 1/sqrt(512)
  const float* st = style + b * SDIM_;
  const float* mw = mod_w + ci * SDIM_;
  float acc = 0.f;
  for (int d = 0; d < SDIM_; ++d) acc += st[d] * mw[d];
  s[t] = acc * smul + mod_b[ci];
}

// ---------------- prep 2: demod[b][co] = rsqrt(sum (scale*w*s)^2 + eps)
__global__ void k_demod(const float* __restrict__ w_conv, const float* __restrict__ s,
                        float* __restrict__ demod) {
  int t = blockIdx.x * blockDim.x + threadIdx.x;
  if (t >= B_ * COUT_) return;
  int b = t / COUT_, co = t % COUT_;
  const float scale = 1.f / 48.f; // 1/sqrt(256*9)
  float acc = 0.f;
  for (int ci = 0; ci < CIN_; ++ci) {
    float sv = s[b * CIN_ + ci] * scale;
    const float* wp = w_conv + (co * CIN_ + ci) * 9;
#pragma unroll
    for (int kk = 0; kk < 9; ++kk) { float w = wp[kk] * sv; acc += w * w; }
  }
  demod[t] = rsqrtf(acc + 1e-8f);
}

// ---------------- prep 3: fold modulate+demod+flip+blur -> bf16 Keff
// layout: keff[(((b*4+par)*9 + dpos)*COUT + co)*CIN + ci]
__global__ void k_keff(const float* __restrict__ w_conv, const float* __restrict__ s,
                       const float* __restrict__ demod, unsigned short* __restrict__ keff) {
  int t = blockIdx.x * blockDim.x + threadIdx.x;
  const int total = B_ * 4 * 9 * COUT_ * CIN_;
  if (t >= total) return;
  int ci   = t & (CIN_ - 1);
  int co   = (t >> 8) & (COUT_ - 1);
  int rest = t >> 15;
  int dpos = rest % 9;
  int par  = (rest / 9) & 3;
  int b    = rest / 36;
  int dy = dpos / 3, dx = dpos % 3;
  int py = par >> 1, px = par & 1;
  // parity tap tables A[p][d][ky] from blur c=[.25,.75,.75,.25]
  const float A0[3][3] = {{0.75f, 0.25f, 0.f}, {0.25f, 0.75f, 0.75f}, {0.f, 0.f, 0.25f}};
  const float A1[3][3] = {{0.25f, 0.f, 0.f}, {0.75f, 0.75f, 0.25f}, {0.f, 0.25f, 0.75f}};
  const float(*Ay)[3] = py ? A1 : A0;
  const float(*Ax)[3] = px ? A1 : A0;
  const float* wp = w_conv + (co * CIN_ + ci) * 9;
  float acc = 0.f;
#pragma unroll
  for (int ky = 0; ky < 3; ++ky) {
    float ay = Ay[dy][ky];
#pragma unroll
    for (int kx = 0; kx < 3; ++kx) {
      float ax = Ax[dx][kx];
      acc += ay * ax * wp[(2 - ky) * 3 + (2 - kx)]; // spatial flip for conv-transpose
    }
  }
  const float scale = 1.f / 48.f;
  float val = acc * scale * s[b * CIN_ + ci] * demod[b * COUT_ + co];
  keff[t] = f2bf(val);
}

// ---------------- prep 4: zero-padded NHWC bf16 copy of x: xt[b][66][66][256]
__global__ void k_xt(const float* __restrict__ x, unsigned short* __restrict__ xt) {
  int t = blockIdx.x * blockDim.x + threadIdx.x;
  const int total = B_ * PROWS * PCOLS * CIN_;
  if (t >= total) return;
  int ci   = t & (CIN_ - 1);
  int rest = t >> 8;
  int c = rest % PCOLS; rest /= PCOLS;
  int r = rest % PROWS;
  int b = rest / PROWS;
  float v = 0.f;
  if (r >= 1 && r <= H_ && c >= 1 && c <= W_)
    v = x[((size_t)(b * CIN_ + ci) * H_ + (r - 1)) * W_ + (c - 1)];
  xt[t] = f2bf(v);
}

// ---------------- main fused implicit-GEMM conv kernel
// block = (b, parity, output row i). 8 waves: wave w -> cout tile w (M=16), 4 N tiles of 16 px.
// Per 32-ci chunk the B operand tile (3x66 px x 32 ci, bf16, [pixel][ci]) is one rectangular
// 3D tile of padded-NHWC xt -> fetched by the Tensor Data Mover straight into LDS,
// double-buffered and pipelined on TENSORcnt.
__global__ __launch_bounds__(256)
void k_conv(const unsigned short* __restrict__ xt, const float* __restrict__ noise,
            const unsigned short* __restrict__ keff, const float* __restrict__ nw,
            const float* __restrict__ act_bias, float* __restrict__ out) {
  __shared__ __align__(32) unsigned short xs[2 * XSBUF]; // two [pixel][ci] bf16 buffers

  const int i    = blockIdx.x;
  const int par  = blockIdx.y;
  const int b    = blockIdx.z;
  const int py   = par >> 1, px = par & 1;
  const int tid  = threadIdx.x;
  const int lane = tid & 31;
  const int wave = tid >> 5;       // mTile 0..7
  const int mrow = lane & 15;      // M row (lane<16) / N column
  const int half = lane >> 4;      // K half selector

  v8f acc[4];
#pragma unroll
  for (int nt = 0; nt < 4; ++nt)
#pragma unroll
    for (int e = 0; e < 8; ++e) acc[nt][e] = 0.f;

  const unsigned short* kbase = keff + (size_t)(b * 4 + par) * 9 * COUT_ * CIN_;
  // chunk source base in padded NHWC: rows i..i+2, col 0, ci = cc*32
  const unsigned short* xrow = xt + (((size_t)b * PROWS + i) * PCOLS) * CIN_;

#ifdef HAVE_TDM
  const unsigned ldsBase = (unsigned)(uintptr_t)(&xs[0]); // low 32 bits = LDS byte offset
  // D# group 1: data_size=2B; tensor dim0=256 (ci), dim1=66 (j); tile 32x66x3;
  // stride0=256, stride1=66*256=16896 (data_size units)
  i32x8 g1;
  g1[0] = (int)(1u << 16);
  g1[1] = (int)(256u << 16);
  g1[2] = (int)(66u << 16);
  g1[3] = (int)(32u << 16);
  g1[4] = (int)(66u | (3u << 16));
  g1[5] = 256;
  g1[6] = (int)(16896u << 16);
  g1[7] = 0;
  i32x4 g2;
  g2[0] = PROWS;                       // tensor_dim2 (i extent)
  g2[1] = 1;                           // tensor_dim3 (unused, tile_dim3=0)
  g2[2] = (int)(PROWS * PCOLS * CIN_); // tensor_dim2_stride low
  g2[3] = 0;                           // stride2 hi | tile_dim3=0
  i32x4 g3;
  g3[0] = 0; g3[1] = 0; g3[2] = 0; g3[3] = 0;
  i32x8 g4;                            // trailing group (6-arg builtin form): unused -> zeros
  g4[0] = 0; g4[1] = 0; g4[2] = 0; g4[3] = 0;
  g4[4] = 0; g4[5] = 0; g4[6] = 0; g4[7] = 0;

  #define TDM_ISSUE(CC)                                                              \
    do {                                                                             \
      unsigned long long ga =                                                        \
          (unsigned long long)(uintptr_t)(xrow + (CC) * CICHUNK);                    \
      u32x4 g0;                                                                      \
      g0[0] = 1u;                                   /* count=1 */                    \
      g0[1] = ldsBase + (unsigned)(((CC) & 1) * (XSBUF * 2));                        \
      g0[2] = (unsigned)ga;                                                          \
      g0[3] = ((unsigned)(ga >> 32) & 0x01FFFFFFu) | (2u << 30); /* type=2 */        \
      __builtin_amdgcn_tensor_load_to_lds(g0, g1, g2, g3, g4, 0);                    \
    } while (0)

  if (wave == 0) TDM_ISSUE(0);
#endif

  for (int cc = 0; cc < NCHUNK; ++cc) {
#ifdef HAVE_TDM
    if (wave == 0) {
      if (cc + 1 < NCHUNK) {
        TDM_ISSUE(cc + 1);                       // next chunk in flight
        __builtin_amdgcn_s_wait_tensorcnt(1);    // current chunk landed
      } else {
        __builtin_amdgcn_s_wait_tensorcnt(0);
      }
    }
    __syncthreads();                             // publish LDS tile to all waves
#else
    __syncthreads();
    // fallback staging: straight 16B copies from padded NHWC xt (no converts, no bounds)
    for (int idx = tid; idx < NROWS * (CICHUNK / 8); idx += 256) {
      int n  = idx >> 2;                 // pixel 0..197
      int wi = idx & 3;                  // 16B chunk within 32-ci row
      int r  = n / XCOLS;
      int c  = n - r * XCOLS;
      const u32x4* src =
          (const u32x4*)(xrow + ((size_t)r * PCOLS + c) * CIN_ + cc * CICHUNK + wi * 8);
      *(u32x4*)(xs + ((cc & 1) * XSBUF) + n * CICHUNK + wi * 8) = *src;
    }
    __syncthreads();
#endif
    const unsigned short* xbuf = xs + (cc & 1) * XSBUF;

#pragma unroll
    for (int dpos = 0; dpos < 9; ++dpos) {
      const int dy = dpos / 3, dx = dpos % 3;
      // A fragment: Keff[co=wave*16+mrow][ci chunk], two contiguous 16B halves
      const unsigned short* kp =
          kbase + ((size_t)dpos * COUT_ + wave * 16 + mrow) * CIN_ + cc * CICHUNK + half * 16;
      AFrag a;
      a.q.lo = *(const u32x4*)kp;
      a.q.hi = *(const u32x4*)(kp + 8);
      if (cc + 1 < NCHUNK) __builtin_prefetch(kp + CICHUNK, 0, 1); // next-chunk Keff
#pragma unroll
      for (int nt = 0; nt < 4; ++nt) {
        int j = nt * 16 + mrow;                 // output col within strip = N column
        int n = dy * XCOLS + j + dx;            // shifted pixel for this tap
        AFrag bf;
        bf.u = *(const u32x8*)(xbuf + n * CICHUNK + half * 16);
        acc[nt] = __builtin_amdgcn_wmma_f32_16x16x32_bf16(
            false, a.bf, false, bf.bf, (short)0, acc[nt], false, false);
      }
    }
#ifdef HAVE_TDM
    __syncthreads();   // all waves done reading before TDM overwrites this buffer
#endif
  }

  // epilogue: + noise*nw, + bias, leaky_relu(0.2) * sqrt(2)
  const float nwv = nw[0];
  const float s2  = 1.4142135623730951f;
  const int oy    = 2 * i + py;
#pragma unroll
  for (int nt = 0; nt < 4; ++nt) {
    int j  = nt * 16 + mrow;
    int ox = 2 * j + px;
    float nz = nwv * noise[((size_t)b * OH_ + oy) * OW_ + ox];
#pragma unroll
    for (int r = 0; r < 8; ++r) {
      int co  = wave * 16 + r + half * 8;       // C/D layout: lane<16 -> M=r, lane>=16 -> M=r+8
      float v = acc[nt][r] + nz + act_bias[co];
      v = (v > 0.f ? v : 0.2f * v) * s2;
      out[(((size_t)b * COUT_ + co) * OH_ + oy) * OW_ + ox] = v;
    }
  }
}

extern "C" void kernel_launch(void* const* d_in, const int* in_sizes, int n_in,
                              void* d_out, int out_size, void* d_ws, size_t ws_size,
                              hipStream_t stream) {
  (void)in_sizes; (void)n_in; (void)out_size; (void)ws_size;
  const float* x        = (const float*)d_in[0];
  const float* style    = (const float*)d_in[1];
  const float* noise    = (const float*)d_in[2];
  const float* w_conv   = (const float*)d_in[3];
  const float* mod_w    = (const float*)d_in[4];
  const float* mod_b    = (const float*)d_in[5];
  const float* nw       = (const float*)d_in[6];
  const float* act_bias = (const float*)d_in[7];
  float* out = (float*)d_out;

  char* ws = (char*)d_ws;
  float* s             = (float*)ws;                    // 4 KB
  float* demod         = (float*)(ws + 4096);           // 2 KB
  unsigned short* keff = (unsigned short*)(ws + 8192);  // 4*4*9*128*256*2 = 9,437,184 B
  unsigned short* xt   = (unsigned short*)(ws + 8192 + (size_t)B_ * 4 * 9 * COUT_ * CIN_ * 2);
                                                        // 4*66*66*256*2 = 8,921,088 B

  k_style<<<dim3((B_ * CIN_ + 255) / 256), dim3(256), 0, stream>>>(style, mod_w, mod_b, s);
  k_demod<<<dim3((B_ * COUT_ + 255) / 256), dim3(256), 0, stream>>>(w_conv, s, demod);
  const int keff_total = B_ * 4 * 9 * COUT_ * CIN_;
  k_keff<<<dim3((keff_total + 255) / 256), dim3(256), 0, stream>>>(w_conv, s, demod, keff);
  const int xt_total = B_ * PROWS * PCOLS * CIN_;
  k_xt<<<dim3((xt_total + 255) / 256), dim3(256), 0, stream>>>(x, xt);
  k_conv<<<dim3(H_, 4, B_), dim3(256), 0, stream>>>(xt, noise, keff, nw, act_bias, out);
}
